// EmformerLayer_77326591197526
// MI455X (gfx1250) — matrix operand order repaired and checked by reading, not compiled, not run on hardware
//
#include <hip/hip_runtime.h>

typedef __bf16 bf16_t;
typedef __attribute__((ext_vector_type(16))) __bf16 v16bf;
typedef __attribute__((ext_vector_type(8)))  __bf16 v8bf;
typedef __attribute__((ext_vector_type(8)))  float  v8f;

// ---- problem constants ----
constexpr int Dc    = 512;
constexpr int Hc    = 8;
constexpr int HDc   = 64;
constexpr int DFFc  = 2048;
constexpr int Uc    = 1024;
constexpr int Bc    = 4;
constexpr int Rc    = 256;
constexpr int Mc    = 32;
constexpr int Sc    = 32;
constexpr int Qc    = 1312;          // R + U + S
constexpr int KVc   = 1312;          // M + R + U
constexpr int PEc   = 2047;
constexpr int PEPc  = 2048;          // padded
constexpr int CHUNKc= 32;
constexpr int KCONV = 31;
constexpr int NTc   = Rc + Uc;       // 1280 seq
constexpr float NEGINF = -100000000.0f;
constexpr float EPSc   = 1e-5f;
constexpr float SCALEc = 0.125f;     // 64^-0.5

// =================== CDNA5 async copy to LDS ===================
// Per-lane 16B gather into LDS; tracked by ASYNCcnt (CDNA5 ISA).
// Generic pointers to __shared__ objects carry the LDS byte offset in addr[31:0]
// (ISA §10.2 LDS aperture), so truncation yields the VDST LDS address.
__device__ inline void async_b128(bf16_t* lds_ptr, const bf16_t* gptr) {
  unsigned loff = (unsigned)(size_t)lds_ptr;
  asm volatile("global_load_async_to_lds_b128 %0, %1, off"
               :: "v"(loff), "v"(gptr) : "memory");
}
__device__ inline void wait_async0() {
  asm volatile("s_wait_asynccnt 0" ::: "memory");
}

// =================== WMMA fragment loaders (wave32) ===================
__device__ inline v16bf pack16(v8bf lo, v8bf hi) {
  v16bf o;
#pragma unroll
  for (int i = 0; i < 8; ++i) { o[i] = lo[i]; o[i + 8] = hi[i]; }
  return o;
}

// A matrix 16x32 bf16, row-major source, K contiguous.
// lane<16: row=lane, K {k0..k0+15}; lane>=16: row=lane-16, K {k0+16..k0+31}
__device__ inline v16bf load_a(const bf16_t* __restrict__ A, int lda,
                               int row0, int k0, int maxr, int lane) {
  int r = row0 + (lane & 15);
  if (r > maxr) r = maxr;
  int kb = k0 + ((lane & 16) ? 16 : 0);
  const bf16_t* p = A + (size_t)r * lda + kb;
  return pack16(*reinterpret_cast<const v8bf*>(p),
                *reinterpret_cast<const v8bf*>(p + 8));
}

// B matrix 32x16 from weight-style W[N,K] row-major (K contiguous per output col).
__device__ inline v16bf load_b_n(const bf16_t* __restrict__ W, int ldb,
                                 int n0, int k0, int maxn, int lane) {
  int n = n0 + (lane & 15);
  if (n > maxn) n = maxn;
  int kb = k0 + ((lane & 16) ? 16 : 0);
  const bf16_t* p = W + (size_t)n * ldb + kb;
  return pack16(*reinterpret_cast<const v8bf*>(p),
                *reinterpret_cast<const v8bf*>(p + 8));
}

// B matrix 32x16 from k-major source V[K,N] (N contiguous) -> strided gathers
__device__ inline v16bf load_b_k(const bf16_t* __restrict__ V, int ldv,
                                 int n0, int k0, int maxk, int lane) {
  int n = n0 + (lane & 15);
  int kb = k0 + ((lane & 16) ? 16 : 0);
  v16bf o;
#pragma unroll
  for (int i = 0; i < 16; ++i) {
    int k = kb + i;
    if (k > maxk) k = maxk;
    o[i] = V[(size_t)k * ldv + n];
  }
  return o;
}

// =================== generic WMMA GEMM ===================
// C[M,N] = Res + scale * act( A[M,K] @ B^T + bias )
// batched over grid.z: z -> (b = z>>3, h = z&7) with per-operand b/h element offsets.
// Non-BKMAJ: 64x128 block tile (wave tile 32x64, 8 wmma / k-step), K double-buffered
// through LDS with async copies. BKMAJ: 64x64 block tile, direct global gathers.
template <int ACT, typename OutT, bool BKMAJ>
__global__ __launch_bounds__(128)
void gemm_kernel(const bf16_t* __restrict__ A, int lda, long long oAb, long long oAh,
                 const bf16_t* __restrict__ Bm, int ldb, long long oBb, long long oBh,
                 const float* __restrict__ bias,
                 OutT* __restrict__ C, int ldc, long long oCb, long long oCh,
                 const float* __restrict__ Res, float scale,
                 int Mr, int N, int Kd) {
  constexpr int NTILES = BKMAJ ? 2 : 4;            // 16-col tiles per wave
  int bz = blockIdx.z;
  int bb = bz >> 3, hh = bz & 7;
  A  += (size_t)bb * oAb + (size_t)hh * oAh;
  Bm += (size_t)bb * oBb + (size_t)hh * oBh;
  size_t coff = (size_t)bb * oCb + (size_t)hh * oCh;

  int wid  = threadIdx.x >> 5;
  int lane = threadIdx.x & 31;
  int wm = wid >> 1, wn = wid & 1;
  int row0 = blockIdx.y * 64 + wm * 32;
  int col0 = blockIdx.x * (32 * NTILES) + wn * (16 * NTILES);

  v8f acc[2][NTILES] = {};

  if constexpr (!BKMAJ) {
    __shared__ bf16_t As[2][64 * 32];
    __shared__ bf16_t Bs[2][128 * 32];
    const int brow0 = blockIdx.y * 64;
    const int bcol0 = blockIdx.x * 128;
    const int t = threadIdx.x;

    auto stage = [&](int buf, int k0) {
#pragma unroll
      for (int e = 0; e < 2; ++e) {                 // A: 64 rows x 4 chunks = 256
        int chunk = t * 2 + e;
        int r = chunk >> 2;
        int h = (chunk & 3) * 8;
        int ra = brow0 + r; if (ra > Mr - 1) ra = Mr - 1;
        async_b128(&As[buf][r * 32 + h], A + (size_t)ra * lda + k0 + h);
      }
#pragma unroll
      for (int e = 0; e < 4; ++e) {                 // B: 128 rows x 4 chunks = 512
        int chunk = t * 4 + e;
        int r = chunk >> 2;
        int h = (chunk & 3) * 8;
        int nb = bcol0 + r; if (nb > N - 1) nb = N - 1;
        async_b128(&Bs[buf][r * 32 + h], Bm + (size_t)nb * ldb + k0 + h);
      }
    };

    const int nk = Kd >> 5;
    stage(0, 0);
    wait_async0();
    __syncthreads();
    for (int kt = 0; kt < nk; ++kt) {
      int cur = kt & 1;
      if (kt + 1 < nk) stage(cur ^ 1, (kt + 1) * 32);   // prefetch next tile (async)
      v16bf a0 = load_a(&As[cur][0], 32, wm * 32,      0, 63, lane);
      v16bf a1 = load_a(&As[cur][0], 32, wm * 32 + 16, 0, 63, lane);
      v16bf bfrag[NTILES];
#pragma unroll
      for (int j = 0; j < NTILES; ++j)
        bfrag[j] = load_b_n(&Bs[cur][0], 32, wn * 64 + j * 16, 0, 127, lane);
#pragma unroll
      for (int j = 0; j < NTILES; ++j) {
        acc[0][j] = __builtin_amdgcn_wmma_f32_16x16x32_bf16(false, a0, false, bfrag[j], (short)0, acc[0][j], false, false);
        acc[1][j] = __builtin_amdgcn_wmma_f32_16x16x32_bf16(false, a1, false, bfrag[j], (short)0, acc[1][j], false, false);
      }
      wait_async0();
      __syncthreads();
    }
  } else {
    for (int k0 = 0; k0 < Kd; k0 += 32) {
      v16bf a0 = load_a(A, lda, row0,      k0, Mr - 1, lane);
      v16bf a1 = load_a(A, lda, row0 + 16, k0, Mr - 1, lane);
      v16bf bfrag[NTILES];
#pragma unroll
      for (int j = 0; j < NTILES; ++j)
        bfrag[j] = load_b_k(Bm, ldb, col0 + j * 16, k0, Kd - 1, lane);
#pragma unroll
      for (int j = 0; j < NTILES; ++j) {
        acc[0][j] = __builtin_amdgcn_wmma_f32_16x16x32_bf16(false, a0, false, bfrag[j], (short)0, acc[0][j], false, false);
        acc[1][j] = __builtin_amdgcn_wmma_f32_16x16x32_bf16(false, a1, false, bfrag[j], (short)0, acc[1][j], false, false);
      }
    }
  }

#pragma unroll
  for (int ti = 0; ti < 2; ++ti) {
#pragma unroll
    for (int tj = 0; tj < NTILES; ++tj) {
      int rbase = row0 + ti * 16 + ((lane < 16) ? 0 : 8);
      int c = col0 + tj * 16 + (lane & 15);
      if (c >= N) continue;
#pragma unroll
      for (int i = 0; i < 8; ++i) {
        int r = rbase + i;
        if (r >= Mr) continue;
        float v = acc[ti][tj][i];
        if (bias) v += bias[c];
        if (ACT == 1) v = v / (1.0f + __expf(-v));  // swish
        size_t idx = coff + (size_t)r * ldc + c;
        if constexpr (sizeof(OutT) == 4) {
          float o = scale * v;
          if (Res) o += Res[idx];
          C[idx] = o;
        } else {
          C[idx] = (OutT)v;
        }
      }
    }
  }
}

// =================== bd (rel-pos) GEMM with rel-shift scatter-add ===================
__global__ __launch_bounds__(128)
void bd_scatter_kernel(const bf16_t* __restrict__ Qv, const bf16_t* __restrict__ P,
                       float* __restrict__ ac) {
  int bz = blockIdx.z;
  int bb = bz >> 3, hh = bz & 7;
  const bf16_t* A = Qv + (size_t)bb * Dc + (size_t)hh * HDc;  // lda = B*D
  const bf16_t* W = P + (size_t)hh * HDc;                      // ldb = D

  int wid  = threadIdx.x >> 5;
  int lane = threadIdx.x & 31;
  int wm = wid >> 1, wn = wid & 1;
  int row0 = blockIdx.y * 64 + wm * 32;   // u
  int col0 = blockIdx.x * 64 + wn * 32;   // pp

  v8f acc[2][2] = {};
  for (int k0 = 0; k0 < HDc; k0 += 32) {
    v16bf a0 = load_a(A, Bc * Dc, row0,      k0, Uc - 1, lane);
    v16bf a1 = load_a(A, Bc * Dc, row0 + 16, k0, Uc - 1, lane);
    v16bf b0 = load_b_n(W, Dc, col0,      k0, PEPc - 1, lane);
    v16bf b1 = load_b_n(W, Dc, col0 + 16, k0, PEPc - 1, lane);
    acc[0][0] = __builtin_amdgcn_wmma_f32_16x16x32_bf16(false, a0, false, b0, (short)0, acc[0][0], false, false);
    acc[0][1] = __builtin_amdgcn_wmma_f32_16x16x32_bf16(false, a0, false, b1, (short)0, acc[0][1], false, false);
    acc[1][0] = __builtin_amdgcn_wmma_f32_16x16x32_bf16(false, a1, false, b0, (short)0, acc[1][0], false, false);
    acc[1][1] = __builtin_amdgcn_wmma_f32_16x16x32_bf16(false, a1, false, b1, (short)0, acc[1][1], false, false);
  }

  float* acb = ac + (size_t)bz * Qc * KVc;
#pragma unroll
  for (int ti = 0; ti < 2; ++ti) {
#pragma unroll
    for (int tj = 0; tj < 2; ++tj) {
      int ubase = row0 + ti * 16 + ((lane < 16) ? 0 : 8);
      int pp = col0 + tj * 16 + (lane & 15);
      if (pp >= PEc) continue;
#pragma unroll
      for (int i = 0; i < 8; ++i) {
        int u = ubase + i;
        int j = pp - (Uc - 1) + u;                 // rel_shift
        if ((unsigned)j < (unsigned)Uc) {
          size_t idx = (size_t)(Rc + u) * KVc + (Mc + Rc + j);
          acb[idx] += acc[ti][tj][i];              // unique (u,j) per (u,pp): no races
        }
      }
    }
  }
}

// =================== elementwise / reduction kernels ===================
__device__ inline float blk_reduce_sum(float v, float* red) {
  int t = threadIdx.x;
  red[t] = v; __syncthreads();
  for (int s = 128; s > 0; s >>= 1) { if (t < s) red[t] += red[t + s]; __syncthreads(); }
  float r = red[0]; __syncthreads();
  return r;
}

__global__ __launch_bounds__(256)
void ln_kernel(const float* __restrict__ X, const float* __restrict__ g,
               const float* __restrict__ be, bf16_t* __restrict__ o1,
               bf16_t* __restrict__ o2, int rows) {
  __shared__ float red[256];
  int row = blockIdx.x;
  if (row >= rows) return;
  const float* x = X + (size_t)row * Dc;
  int t = threadIdx.x;
  float a = x[t], b = x[t + 256];
  float mean = blk_reduce_sum(a + b, red) * (1.0f / Dc);
  float da = a - mean, db = b - mean;
  float var = blk_reduce_sum(da * da + db * db, red) * (1.0f / Dc);
  float inv = rsqrtf(var + EPSc);
  float ya = da * inv * g[t] + be[t];
  float yb = db * inv * g[t + 256] + be[t + 256];
  size_t base = (size_t)row * Dc;
  o1[base + t] = (bf16_t)ya; o1[base + t + 256] = (bf16_t)yb;
  if (o2) { o2[base + t] = (bf16_t)ya; o2[base + t + 256] = (bf16_t)yb; }
}

__global__ __launch_bounds__(256)
void summary_kernel(bf16_t* __restrict__ qin) {
  int cb = blockIdx.x;                 // c*B + b
  int c = cb / Bc, b = cb % Bc;
  for (int e = threadIdx.x; e < Dc; e += 256) {
    float s = 0.0f;
    for (int i = 0; i < CHUNKc; ++i) {
      int u = c * CHUNKc + i;
      s += (float)qin[((size_t)(Rc + u) * Bc + b) * Dc + e];
    }
    qin[((size_t)(NTc + c) * Bc + b) * Dc + e] = (bf16_t)(s * (1.0f / CHUNKc));
  }
}

__global__ void qprep_kernel(const bf16_t* __restrict__ qb,
                             const float* __restrict__ pbu,
                             const float* __restrict__ pbv,
                             bf16_t* __restrict__ qu, bf16_t* __restrict__ qv) {
  size_t idx = (size_t)blockIdx.x * 256 + threadIdx.x;
  if (idx >= (size_t)Qc * Bc * Dc) return;
  int c = (int)(idx % Dc);
  size_t row = idx / Dc;
  float v = (float)qb[idx];
  qu[idx] = (bf16_t)(v + pbu[c]);
  if (row >= (size_t)Rc * Bc && row < (size_t)NTc * Bc)
    qv[(row - (size_t)Rc * Bc) * Dc + c] = (bf16_t)(v + pbv[c]);
}

__global__ __launch_bounds__(256)
void softmax_kernel(float* __restrict__ ac, const unsigned char* __restrict__ mask,
                    const int* __restrict__ lengths) {
  __shared__ float red[256];
  int rowid = blockIdx.x;              // bh*Q + qi
  int qi = rowid % Qc;
  int bh = rowid / Qc;
  int b = bh >> 3;
  float* rowp = ac + (size_t)rowid * KVc;
  const unsigned char* mrow = mask + (size_t)qi * KVc;
  int lim = Mc + Rc + lengths[b];
  int t = threadIdx.x;
  float vals[6];
  float mx = -3.4e38f;
#pragma unroll
  for (int i = 0; i < 6; ++i) {
    int k = t + i * 256;
    float w = NEGINF;
    if (k < KVc) {
      w = rowp[k] * SCALEc;
      if (mrow[k] || k >= lim) w = NEGINF;
    }
    vals[i] = w;
    mx = fmaxf(mx, w);
  }
  red[t] = mx; __syncthreads();
  for (int s = 128; s > 0; s >>= 1) { if (t < s) red[t] = fmaxf(red[t], red[t + s]); __syncthreads(); }
  mx = red[0]; __syncthreads();
  float sum = 0.0f;
#pragma unroll
  for (int i = 0; i < 6; ++i) {
    int k = t + i * 256;
    if (k < KVc) { float e = __expf(vals[i] - mx); vals[i] = e; sum += e; }
  }
  sum = blk_reduce_sum(sum, red);
  float inv = 1.0f / sum;
  bf16_t* op = reinterpret_cast<bf16_t*>(rowp);   // probs bf16 in-place (reads done)
#pragma unroll
  for (int i = 0; i < 6; ++i) {
    int k = t + i * 256;
    if (k < KVc) op[k] = (bf16_t)(vals[i] * inv);
  }
}

__global__ void attn_resid_kernel(const float* __restrict__ ob, float* __restrict__ src,
                                  float* __restrict__ out_mem) {
  size_t idx = (size_t)blockIdx.x * 256 + threadIdx.x;
  if (idx >= (size_t)Qc * Bc * Dc) return;
  size_t row = idx / Dc;
  if (row < (size_t)NTc * Bc) {
    src[idx] += ob[idx];
  } else {
    float v = ob[idx];
    out_mem[idx - (size_t)NTc * Bc * Dc] = fminf(10.0f, fmaxf(-10.0f, v));
  }
}

__global__ void glu_kernel(const bf16_t* __restrict__ y1, bf16_t* __restrict__ g) {
  size_t idx = (size_t)blockIdx.x * 256 + threadIdx.x;
  if (idx >= (size_t)Uc * Bc * Dc) return;
  size_t row = idx / Dc;
  int c = (int)(idx % Dc);
  float a  = (float)y1[row * (2 * Dc) + c];
  float bg = (float)y1[row * (2 * Dc) + Dc + c];
  g[idx] = (bf16_t)(a / (1.0f + __expf(-bg)));
}

__global__ __launch_bounds__(256)
void dwconv_ln_swish_kernel(const bf16_t* __restrict__ g, const float* __restrict__ dww,
                            const float* __restrict__ dwb, const float* __restrict__ cg,
                            const float* __restrict__ cb, bf16_t* __restrict__ out) {
  __shared__ float red[256];
  int r = blockIdx.x;                  // u*B + b
  int u = r / Bc, b = r % Bc;
  int t = threadIdx.x;
  float z[2];
#pragma unroll
  for (int e = 0; e < 2; ++e) {
    int c = t + e * 256;
    float acc = dwb[c];
#pragma unroll
    for (int j = 0; j < KCONV; ++j) {
      int up = u - (KCONV - 1) + j;
      if (up >= 0) acc += (float)g[((size_t)up * Bc + b) * Dc + c] * dww[c * KCONV + j];
    }
    z[e] = acc;
  }
  float mean = blk_reduce_sum(z[0] + z[1], red) * (1.0f / Dc);
  float d0 = z[0] - mean, d1 = z[1] - mean;
  float var = blk_reduce_sum(d0 * d0 + d1 * d1, red) * (1.0f / Dc);
  float inv = rsqrtf(var + EPSc);
  float s0 = d0 * inv * cg[t] + cb[t];
  float s1 = d1 * inv * cg[t + 256] + cb[t + 256];
  out[(size_t)r * Dc + t]       = (bf16_t)(s0 / (1.0f + __expf(-s0)));
  out[(size_t)r * Dc + t + 256] = (bf16_t)(s1 / (1.0f + __expf(-s1)));
}

__global__ __launch_bounds__(256)
void final_ln_kernel(const float* __restrict__ src, const float* __restrict__ g,
                     const float* __restrict__ be, float* __restrict__ out) {
  __shared__ float red[256];
  int row = blockIdx.x;                // seq*B + b
  const float* x = src + (size_t)row * Dc;
  int t = threadIdx.x;
  float a = x[t], b2 = x[t + 256];
  float mean = blk_reduce_sum(a + b2, red) * (1.0f / Dc);
  float da = a - mean, db = b2 - mean;
  float var = blk_reduce_sum(da * da + db * db, red) * (1.0f / Dc);
  float inv = rsqrtf(var + EPSc);
  float ya = da * inv * g[t] + be[t];
  float yb = db * inv * g[t + 256] + be[t + 256];
  int seq = row / Bc, b = row % Bc;
  size_t dst;
  if (seq >= Rc) dst = ((size_t)(seq - Rc) * Bc + b) * Dc;                         // utterance out
  else           dst = (size_t)Uc * Bc * Dc + ((size_t)seq * Bc + b) * Dc;         // right-context out
  out[dst + t] = ya;
  out[dst + t + 256] = yb;
}

__global__ void castf2b_kernel(bf16_t* __restrict__ dst, const float* __restrict__ src, size_t n) {
  size_t i = (size_t)blockIdx.x * 256 + threadIdx.x;
  if (i < n) dst[i] = (bf16_t)src[i];
}

__global__ void cast_pospad_kernel(bf16_t* __restrict__ dst, const float* __restrict__ src) {
  size_t i = (size_t)blockIdx.x * 256 + threadIdx.x;
  if (i >= (size_t)PEPc * Dc) return;
  size_t row = i / Dc;
  dst[i] = (row < (size_t)PEc) ? (bf16_t)src[i] : (bf16_t)0.0f;
}

// =================== host launcher ===================
static inline unsigned cdivu(long long a, long long b) { return (unsigned)((a + b - 1) / b); }

extern "C" void kernel_launch(void* const* d_in, const int* in_sizes, int n_in,
                              void* d_out, int out_size, void* d_ws, size_t ws_size,
                              hipStream_t stream) {
  const float* in_utt  = (const float*)d_in[0];
  const float* in_rc   = (const float*)d_in[1];
  const float* in_mem  = (const float*)d_in[2];
  const float* in_pos  = (const float*)d_in[3];
  const int*   in_len  = (const int*)d_in[4];
  const unsigned char* in_mask = (const unsigned char*)d_in[5];
  const float* Wq   = (const float*)d_in[6];  const float* bq   = (const float*)d_in[7];
  const float* Wkv  = (const float*)d_in[8];  const float* bkv  = (const float*)d_in[9];
  const float* Wout = (const float*)d_in[10]; const float* bout = (const float*)d_in[11];
  const float* Wpos = (const float*)d_in[12];
  const float* pbu  = (const float*)d_in[13]; const float* pbv  = (const float*)d_in[14];
  const float* Wm1  = (const float*)d_in[15]; const float* bm1  = (const float*)d_in[16];
  const float* Wm2  = (const float*)d_in[17]; const float* bm2  = (const float*)d_in[18];
  const float* Wf1  = (const float*)d_in[19]; const float* bf1  = (const float*)d_in[20];
  const float* Wf2  = (const float*)d_in[21]; const float* bf2  = (const float*)d_in[22];
  const float* pw1w = (const float*)d_in[23]; const float* pw1b = (const float*)d_in[24];
  const float* dww  = (const float*)d_in[25]; const float* dwb  = (const float*)d_in[26];
  const float* clng = (const float*)d_in[27]; const float* clnb = (const float*)d_in[28];
  const float* pw2w = (const float*)d_in[29]; const float* pw2b = (const float*)d_in[30];
  const float* lnmg = (const float*)d_in[31]; const float* lnmb = (const float*)d_in[32];
  const float* lnag = (const float*)d_in[33]; const float* lnab = (const float*)d_in[34];
  const float* lncg = (const float*)d_in[35]; const float* lncb = (const float*)d_in[36];
  const float* lnfg = (const float*)d_in[37]; const float* lnfb = (const float*)d_in[38];
  const float* lnog = (const float*)d_in[39]; const float* lnob = (const float*)d_in[40];
  (void)in_sizes; (void)n_in; (void)out_size; (void)ws_size;

  // ---- workspace carve-up ----
  size_t off = 0;
  auto alloc = [&](size_t bytes) -> char* {
    char* p = (char*)d_ws + off;
    off += (bytes + 255) & ~(size_t)255;
    return p;
  };
  const size_t NTB = (size_t)NTc * Bc;     // 5120 rows
  const size_t QB  = (size_t)Qc * Bc;      // 5248 rows
  const size_t UB  = (size_t)Uc * Bc;      // 4096 rows

  float*  src   = (float*)alloc(NTB * Dc * 4);
  bf16_t* xa    = (bf16_t*)alloc(NTB * Dc * 2);          // LN staging (reused)
  bf16_t* h1    = (bf16_t*)alloc(NTB * DFFc * 2);        // FFN hidden / pw1 out (reused)
  bf16_t* qin   = (bf16_t*)alloc(QB * Dc * 2);
  bf16_t* kvin  = (bf16_t*)alloc(QB * Dc * 2);
  bf16_t* qbuf  = (bf16_t*)alloc(QB * Dc * 2);
  bf16_t* qu    = (bf16_t*)alloc(QB * Dc * 2);
  bf16_t* qv    = (bf16_t*)alloc(UB * Dc * 2);
  bf16_t* kvb   = (bf16_t*)alloc(QB * (2 * Dc) * 2);
  bf16_t* posin = (bf16_t*)alloc((size_t)PEPc * Dc * 2);
  bf16_t* pb    = (bf16_t*)alloc((size_t)PEPc * Dc * 2);
  bf16_t* attn  = (bf16_t*)alloc(QB * Dc * 2);
  float*  obuf  = (float*)alloc(QB * Dc * 4);
  bf16_t* gg    = (bf16_t*)alloc(UB * Dc * 2);
  float*  ac    = (float*)alloc((size_t)Bc * Hc * Qc * KVc * 4);
  bf16_t* wWm1  = (bf16_t*)alloc((size_t)DFFc * Dc * 2);
  bf16_t* wWm2  = (bf16_t*)alloc((size_t)Dc * DFFc * 2);
  bf16_t* wWf1  = (bf16_t*)alloc((size_t)DFFc * Dc * 2);
  bf16_t* wWf2  = (bf16_t*)alloc((size_t)Dc * DFFc * 2);
  bf16_t* wWq   = (bf16_t*)alloc((size_t)Dc * Dc * 2);
  bf16_t* wWkv  = (bf16_t*)alloc((size_t)2 * Dc * Dc * 2);
  bf16_t* wWout = (bf16_t*)alloc((size_t)Dc * Dc * 2);
  bf16_t* wWpos = (bf16_t*)alloc((size_t)Dc * Dc * 2);
  bf16_t* wpw1  = (bf16_t*)alloc((size_t)2 * Dc * Dc * 2);
  bf16_t* wpw2  = (bf16_t*)alloc((size_t)Dc * Dc * 2);

  auto castw = [&](bf16_t* dst, const float* s, size_t n) {
    castf2b_kernel<<<cdivu((long long)n, 256), 256, 0, stream>>>(dst, s, n);
  };
  castw(wWm1, Wm1, (size_t)DFFc * Dc);
  castw(wWm2, Wm2, (size_t)Dc * DFFc);
  castw(wWf1, Wf1, (size_t)DFFc * Dc);
  castw(wWf2, Wf2, (size_t)Dc * DFFc);
  castw(wWq,  Wq,  (size_t)Dc * Dc);
  castw(wWkv, Wkv, (size_t)2 * Dc * Dc);
  castw(wWout,Wout,(size_t)Dc * Dc);
  castw(wWpos,Wpos,(size_t)Dc * Dc);
  castw(wpw1, pw1w,(size_t)2 * Dc * Dc);
  castw(wpw2, pw2w,(size_t)Dc * Dc);
  cast_pospad_kernel<<<cdivu((long long)PEPc * Dc, 256), 256, 0, stream>>>(posin, in_pos);
  castw(kvin, in_mem, (size_t)Mc * Bc * Dc);   // kv seq 0..31 = memory

  // src = concat([right_context, utterance])
  hipMemcpyAsync(src, in_rc, (size_t)Rc * Bc * Dc * 4, hipMemcpyDeviceToDevice, stream);
  hipMemcpyAsync(src + (size_t)Rc * Bc * Dc, in_utt, (size_t)Uc * Bc * Dc * 4,
                 hipMemcpyDeviceToDevice, stream);

  // ---- macaron FFN: src += 0.5*W2(swish(W1(ln(src)))) ----
  ln_kernel<<<(unsigned)NTB, 256, 0, stream>>>(src, lnmg, lnmb, xa, nullptr, (int)NTB);
  gemm_kernel<1, bf16_t, false><<<dim3(cdivu(DFFc,128), cdivu(NTB,64), 1), 128, 0, stream>>>(
      xa, Dc, 0, 0, wWm1, Dc, 0, 0, bm1, h1, DFFc, 0, 0, nullptr, 1.0f, (int)NTB, DFFc, Dc);
  gemm_kernel<0, float, false><<<dim3(cdivu(Dc,128), cdivu(NTB,64), 1), 128, 0, stream>>>(
      h1, DFFc, 0, 0, wWm2, DFFc, 0, 0, bm2, src, Dc, 0, 0, src, 0.5f, (int)NTB, Dc, DFFc);

  // ---- attention ----
  ln_kernel<<<(unsigned)NTB, 256, 0, stream>>>(src, lnag, lnab, qin,
                                               kvin + (size_t)Mc * Bc * Dc, (int)NTB);
  summary_kernel<<<Sc * Bc, 256, 0, stream>>>(qin);
  gemm_kernel<0, bf16_t, false><<<dim3(cdivu(Dc,128), cdivu(QB,64), 1), 128, 0, stream>>>(
      qin, Dc, 0, 0, wWq, Dc, 0, 0, bq, qbuf, Dc, 0, 0, nullptr, 1.0f, (int)QB, Dc, Dc);
  qprep_kernel<<<cdivu((long long)QB * Dc, 256), 256, 0, stream>>>(qbuf, pbu, pbv, qu, qv);
  gemm_kernel<0, bf16_t, false><<<dim3(cdivu(2*Dc,128), cdivu(QB,64), 1), 128, 0, stream>>>(
      kvin, Dc, 0, 0, wWkv, Dc, 0, 0, bkv, kvb, 2*Dc, 0, 0, nullptr, 1.0f, (int)QB, 2*Dc, Dc);
  gemm_kernel<0, bf16_t, false><<<dim3(cdivu(Dc,128), cdivu(PEPc,64), 1), 128, 0, stream>>>(
      posin, Dc, 0, 0, wWpos, Dc, 0, 0, nullptr, pb, Dc, 0, 0, nullptr, 1.0f, PEPc, Dc, Dc);

  // scores: ac[b,h,q,k] = (q+pbu) . key    (batched over 32 = B*H)
  gemm_kernel<0, float, false><<<dim3(cdivu(KVc,128), cdivu(Qc,64), Bc*Hc), 128, 0, stream>>>(
      qu, Bc*Dc, Dc, HDc,
      kvb, Bc*2*Dc, 2*Dc, HDc,
      nullptr,
      ac, KVc, (long long)Hc*Qc*KVc, (long long)Qc*KVc,
      nullptr, 1.0f, Qc, KVc, HDc);

  // bd: (q_utt + pbv) . p, rel-shifted, added into ac
  bd_scatter_kernel<<<dim3(cdivu(PEPc,64), cdivu(Uc,64), Bc*Hc), 128, 0, stream>>>(qv, pb, ac);

  // softmax with masks; probs written bf16 in-place over each fp32 row
  softmax_kernel<<<(unsigned)(Bc * Hc * Qc), 256, 0, stream>>>(ac, in_mask, in_len);

  // attn = probs @ value  (batched, B k-major), output staged bf16 as (q*B+b, h*64+hd)
  gemm_kernel<0, bf16_t, true><<<dim3(1, cdivu(Qc,64), Bc*Hc), 128, 0, stream>>>(
      (const bf16_t*)ac, 2*KVc, (long long)Hc*Qc*2*KVc, (long long)Qc*2*KVc,
      kvb + Dc, Bc*2*Dc, 2*Dc, HDc,
      nullptr,
      attn, Bc*Dc, Dc, HDc,
      nullptr, 1.0f, Qc, HDc, KVc);

  // out projection + split into residual / out_mem
  gemm_kernel<0, float, false><<<dim3(cdivu(Dc,128), cdivu(QB,64), 1), 128, 0, stream>>>(
      attn, Dc, 0, 0, wWout, Dc, 0, 0, bout, obuf, Dc, 0, 0, nullptr, 1.0f, (int)QB, Dc, Dc);
  attn_resid_kernel<<<cdivu((long long)QB * Dc, 256), 256, 0, stream>>>(
      obuf, src, (float*)d_out + (size_t)(Uc + Rc) * Bc * Dc);

  // ---- conv module on utterance rows ----
  ln_kernel<<<(unsigned)UB, 256, 0, stream>>>(src + (size_t)Rc * Bc * Dc, lncg, lncb,
                                              xa, nullptr, (int)UB);
  gemm_kernel<0, bf16_t, false><<<dim3(cdivu(2*Dc,128), cdivu(UB,64), 1), 128, 0, stream>>>(
      xa, Dc, 0, 0, wpw1, Dc, 0, 0, pw1b, h1, 2*Dc, 0, 0, nullptr, 1.0f, (int)UB, 2*Dc, Dc);
  glu_kernel<<<cdivu((long long)UB * Dc, 256), 256, 0, stream>>>(h1, gg);
  dwconv_ln_swish_kernel<<<(unsigned)UB, 256, 0, stream>>>(gg, dww, dwb, clng, clnb, xa);
  gemm_kernel<0, float, false><<<dim3(cdivu(Dc,128), cdivu(UB,64), 1), 128, 0, stream>>>(
      xa, Dc, 0, 0, wpw2, Dc, 0, 0, pw2b,
      src + (size_t)Rc * Bc * Dc, Dc, 0, 0,
      src + (size_t)Rc * Bc * Dc, 1.0f, (int)UB, Dc, Dc);

  // ---- final FFN ----
  ln_kernel<<<(unsigned)NTB, 256, 0, stream>>>(src, lnfg, lnfb, xa, nullptr, (int)NTB);
  gemm_kernel<1, bf16_t, false><<<dim3(cdivu(DFFc,128), cdivu(NTB,64), 1), 128, 0, stream>>>(
      xa, Dc, 0, 0, wWf1, Dc, 0, 0, bf1, h1, DFFc, 0, 0, nullptr, 1.0f, (int)NTB, DFFc, Dc);
  gemm_kernel<0, float, false><<<dim3(cdivu(Dc,128), cdivu(NTB,64), 1), 128, 0, stream>>>(
      h1, DFFc, 0, 0, wWf2, DFFc, 0, 0, bf2, src, Dc, 0, 0, src, 0.5f, (int)NTB, Dc, DFFc);

  // ---- final LN, write (utterance, right_context) outputs ----
  final_ln_kernel<<<(unsigned)NTB, 256, 0, stream>>>(src, lnog, lnob, (float*)d_out);
}
